// Attention_80341658239433
// MI455X (gfx1250) — compile-verified, matrix operand order checked by reading, not run on hardware
//
#include <hip/hip_runtime.h>
#include <hip/hip_bf16.h>
#include <math.h>

// ---------------------------------------------------------------------------
// Fused LN + MHA + out-proj for MI455X (gfx1250), wave32 + WMMA f16.
//   B=4, N=2048, DIM=512, H=8, D=64, INNER=512, scale = 1/8
// ---------------------------------------------------------------------------

#define B_   4
#define N_   2048
#define DIM_ 512
#define H_   8
#define D_   64
#define ROWS (B_ * N_)          // 8192
#define QKVN (3 * DIM_)         // 1536
#define SCALE_ 0.125f
#define EPS_ 1e-5f

typedef __attribute__((ext_vector_type(4)))  _Float16 v4h;
typedef __attribute__((ext_vector_type(8)))  _Float16 v8h;
typedef __attribute__((ext_vector_type(16))) _Float16 v16h;
typedef __attribute__((ext_vector_type(8)))  float    v8f;

// Load one WMMA 16x32 f16 fragment (A or B) when the K dimension is
// contiguous in memory. Caller passes p = row_base + k0 + hi*8; the lane's
// 16 halves are [p..p+7] and [p+16..p+23] per the CDNA5 16-bit layout.
__device__ __forceinline__ v16h load_frag_k(const _Float16* p) {
  v8h a = *(const v8h*)(p);
  v8h b = *(const v8h*)(p + 16);
  return __builtin_shufflevector(a, b, 0,1,2,3,4,5,6,7,8,9,10,11,12,13,14,15);
}

__device__ __forceinline__ v8f wmma_f16(v16h a, v16h b, v8f c) {
  return __builtin_amdgcn_wmma_f32_16x16x32_f16(false, a, false, b,
                                                (short)0, c, false, false);
}

// ---------------------------------------------------------------------------
// Kernel 1: cast qkv_w (1536x512) and out_w (512x512) f32 -> f16
// ---------------------------------------------------------------------------
__global__ __launch_bounds__(256) void cast_weights_k(
    const float* __restrict__ qkv_w, const float* __restrict__ out_w,
    _Float16* __restrict__ wq, _Float16* __restrict__ wo) {
  const unsigned i = blockIdx.x * 256u + threadIdx.x;   // float4 index
  const unsigned NQ4 = (QKVN * DIM_) / 4;               // 196608
  if (i < NQ4) {
    float4 v = ((const float4*)qkv_w)[i];
    v4h h = { (_Float16)v.x, (_Float16)v.y, (_Float16)v.z, (_Float16)v.w };
    *(v4h*)(wq + (size_t)i * 4) = h;
  } else {
    unsigned j = i - NQ4;                               // < 65536
    float4 v = ((const float4*)out_w)[j];
    v4h h = { (_Float16)v.x, (_Float16)v.y, (_Float16)v.z, (_Float16)v.w };
    *(v4h*)(wo + (size_t)j * 4) = h;
  }
}

// ---------------------------------------------------------------------------
// Kernel 2: LayerNorm over DIM=512, output f16. One 128-thread block per row.
// ---------------------------------------------------------------------------
__global__ __launch_bounds__(128) void layernorm_k(
    const float* __restrict__ x, const float* __restrict__ w,
    const float* __restrict__ b, _Float16* __restrict__ xn) {
  __shared__ float red[128];
  const unsigned row = blockIdx.x, tid = threadIdx.x;
  const float4 xv = ((const float4*)(x + (size_t)row * DIM_))[tid];

  float s = xv.x + xv.y + xv.z + xv.w;
  red[tid] = s; __syncthreads();
  for (int off = 64; off > 0; off >>= 1) {
    if (tid < (unsigned)off) red[tid] += red[tid + off];
    __syncthreads();
  }
  const float mu = red[0] * (1.0f / DIM_);
  __syncthreads();

  float dx0 = xv.x - mu, dx1 = xv.y - mu, dx2 = xv.z - mu, dx3 = xv.w - mu;
  red[tid] = dx0*dx0 + dx1*dx1 + dx2*dx2 + dx3*dx3; __syncthreads();
  for (int off = 64; off > 0; off >>= 1) {
    if (tid < (unsigned)off) red[tid] += red[tid + off];
    __syncthreads();
  }
  const float rstd = rsqrtf(red[0] * (1.0f / DIM_) + EPS_);

  const float4 wv = ((const float4*)w)[tid];
  const float4 bv = ((const float4*)b)[tid];
  v4h h = { (_Float16)(dx0 * rstd * wv.x + bv.x),
            (_Float16)(dx1 * rstd * wv.y + bv.y),
            (_Float16)(dx2 * rstd * wv.z + bv.z),
            (_Float16)(dx3 * rstd * wv.w + bv.w) };
  *(v4h*)(xn + (size_t)row * DIM_ + tid * 4) = h;
}

// ---------------------------------------------------------------------------
// Kernel 3: QKV GEMM  C[8192,1536] = xn[8192,512] @ qkv_w[1536,512]^T + bias.
// NT form: both operands contiguous along K. Per-wave 16x64 tile;
// block = 8 waves = 32(M) x 256(N). Epilogue scatters into Q/K [bh,n,d] and
// V transposed [bh,d,n] (f16) so attention's PV fragments load contiguously.
// ---------------------------------------------------------------------------
__global__ __launch_bounds__(256) void qkv_gemm_k(
    const _Float16* __restrict__ xn, const _Float16* __restrict__ w,
    const float* __restrict__ bias, _Float16* __restrict__ q,
    _Float16* __restrict__ k, _Float16* __restrict__ vt) {
  const unsigned tid = threadIdx.x, wave = tid >> 5, lane = tid & 31;
  const unsigned lo = lane & 15, hi = lane >> 4;
  const unsigned m0 = blockIdx.y * 32 + (wave >> 2) * 16;
  const unsigned n0 = blockIdx.x * 256 + (wave & 3) * 64;

  const _Float16* arow = xn + (size_t)(m0 + lo) * DIM_ + hi * 8;
  v8f acc[4] = {};
  for (int k0 = 0; k0 < DIM_; k0 += 32) {
    __builtin_prefetch(arow + k0 + 128, 0, 0);
    v16h af = load_frag_k(arow + k0);
#pragma unroll
    for (int f = 0; f < 4; ++f) {
      const _Float16* brow = w + (size_t)(n0 + f * 16 + lo) * DIM_ + hi * 8;
      v16h bf = load_frag_k(brow + k0);
      acc[f] = wmma_f16(af, bf, acc[f]);
    }
  }
#pragma unroll
  for (int f = 0; f < 4; ++f) {
    const unsigned col = n0 + f * 16 + lo;
    const float bv = bias[col];
#pragma unroll
    for (int r = 0; r < 8; ++r) {
      const unsigned row = m0 + r + 8 * hi;
      const _Float16 hv = (_Float16)(acc[f][r] + bv);
      const unsigned bb = row >> 11, nn = row & (N_ - 1);
      if (col < DIM_) {
        const unsigned h = col >> 6, d = col & 63;
        q[(((size_t)(bb * H_ + h)) * N_ + nn) * D_ + d] = hv;
      } else if (col < 2 * DIM_) {
        const unsigned c = col - DIM_, h = c >> 6, d = c & 63;
        k[(((size_t)(bb * H_ + h)) * N_ + nn) * D_ + d] = hv;
      } else {
        const unsigned c = col - 2 * DIM_, h = c >> 6, d = c & 63;
        vt[(((size_t)(bb * H_ + h)) * D_ + d) * N_ + nn] = hv;
      }
    }
  }
}

// ---------------------------------------------------------------------------
// Kernel 4: Flash attention. Block = 8 waves; each wave owns a 16-row Q tile
// (Q fragments live in registers the whole time), streams KV in 32-col
// chunks. Online softmax row stats live in the C-layout (vgpr r <-> row
// r+8*hi, lane lo <-> col). P goes through LDS to re-shape C-layout -> A-frag.
// ---------------------------------------------------------------------------
__global__ __launch_bounds__(256) void attention_k(
    const _Float16* __restrict__ q, const _Float16* __restrict__ kk,
    const _Float16* __restrict__ vt, _Float16* __restrict__ o) {
  __shared__ _Float16 plds[8][16 * 32 + 16];
  const unsigned tid = threadIdx.x, wave = tid >> 5, lane = tid & 31;
  const unsigned lo = lane & 15, hi = lane >> 4;
  const unsigned bh = blockIdx.x >> 4;          // 0..31
  const unsigned qg = blockIdx.x & 15;          // 0..15
  const unsigned b = bh >> 3, h = bh & 7;
  const unsigned q0 = qg * 128 + wave * 16;

  const _Float16* qbase = q  + (size_t)bh * N_ * D_;
  const _Float16* kbase = kk + (size_t)bh * N_ * D_;
  const _Float16* vbase = vt + (size_t)bh * D_ * N_;

  const _Float16* qrow = qbase + (size_t)(q0 + lo) * D_ + hi * 8;
  const v16h qf0 = load_frag_k(qrow);        // d = 0..31
  const v16h qf1 = load_frag_k(qrow + 32);   // d = 32..63

  float m[8], l[8];
  v8f acc[4] = {};
#pragma unroll
  for (int r = 0; r < 8; ++r) { m[r] = -INFINITY; l[r] = 0.0f; }
  _Float16* pt = &plds[wave][0];

  for (int j0 = 0; j0 < N_; j0 += 32) {
    // ---- S = Q K^T for two 16-col tiles (k = 64 split into 2x32) ----
    v8f s0 = {}, s1 = {};
    {
      const _Float16* kr = kbase + (size_t)(j0 + lo) * D_ + hi * 8;
      s0 = wmma_f16(qf0, load_frag_k(kr), s0);
      s0 = wmma_f16(qf1, load_frag_k(kr + 32), s0);
      const _Float16* kr2 = kbase + (size_t)(j0 + 16 + lo) * D_ + hi * 8;
      s1 = wmma_f16(qf0, load_frag_k(kr2), s1);
      s1 = wmma_f16(qf1, load_frag_k(kr2 + 32), s1);
    }
    // ---- online softmax over the 32 columns ----
#pragma unroll
    for (int r = 0; r < 8; ++r) {
      const float a0 = s0[r] * SCALE_, a1 = s1[r] * SCALE_;
      float mx = fmaxf(a0, a1);
      mx = fmaxf(mx, __shfl_xor(mx, 1, 16));
      mx = fmaxf(mx, __shfl_xor(mx, 2, 16));
      mx = fmaxf(mx, __shfl_xor(mx, 4, 16));
      mx = fmaxf(mx, __shfl_xor(mx, 8, 16));
      const float mn = fmaxf(m[r], mx);
      const float alpha = __expf(m[r] - mn);
      const float p0 = __expf(a0 - mn), p1 = __expf(a1 - mn);
      float rs = p0 + p1;
      rs += __shfl_xor(rs, 1, 16);
      rs += __shfl_xor(rs, 2, 16);
      rs += __shfl_xor(rs, 4, 16);
      rs += __shfl_xor(rs, 8, 16);
      l[r] = l[r] * alpha + rs;
      m[r] = mn;
#pragma unroll
      for (int f = 0; f < 4; ++f) acc[f][r] *= alpha;
      const unsigned rr = r + 8 * hi;          // C-layout row
      pt[rr * 32 + lo]      = (_Float16)p0;
      pt[rr * 32 + 16 + lo] = (_Float16)p1;
    }
    // intra-wave: DS writes above feed DS reads below (cross-lane)
    asm volatile("s_wait_dscnt 0" ::: "memory");
    const v16h pf = load_frag_k(pt + lo * 32 + hi * 8);  // 16x32 A-frag
    // ---- O += P V : V^T gives contiguous-K B fragments ----
#pragma unroll
    for (int f = 0; f < 4; ++f) {
      const _Float16* vr = vbase + (size_t)(f * 16 + lo) * N_ + j0 + hi * 8;
      acc[f] = wmma_f16(pf, load_frag_k(vr), acc[f]);
    }
  }

  // ---- normalize and store O (f16) into [b, n, h*64 + d] ----
  _Float16* ob = o + ((size_t)(b * N_ + q0)) * DIM_ + h * D_;
#pragma unroll
  for (int r = 0; r < 8; ++r) {
    const float inv = 1.0f / l[r];
    const unsigned rr = r + 8 * hi;
#pragma unroll
    for (int f = 0; f < 4; ++f)
      ob[(size_t)rr * DIM_ + f * 16 + lo] = (_Float16)(acc[f][r] * inv);
  }
}

// ---------------------------------------------------------------------------
// Kernel 5: output projection  out[8192,512] = O[8192,512] @ out_w^T + out_b
// ---------------------------------------------------------------------------
__global__ __launch_bounds__(256) void out_gemm_k(
    const _Float16* __restrict__ oin, const _Float16* __restrict__ w,
    const float* __restrict__ bias, float* __restrict__ out) {
  const unsigned tid = threadIdx.x, wave = tid >> 5, lane = tid & 31;
  const unsigned lo = lane & 15, hi = lane >> 4;
  const unsigned m0 = blockIdx.y * 32 + (wave >> 2) * 16;
  const unsigned n0 = blockIdx.x * 256 + (wave & 3) * 64;

  const _Float16* arow = oin + (size_t)(m0 + lo) * DIM_ + hi * 8;
  v8f acc[4] = {};
  for (int k0 = 0; k0 < DIM_; k0 += 32) {
    __builtin_prefetch(arow + k0 + 128, 0, 0);
    v16h af = load_frag_k(arow + k0);
#pragma unroll
    for (int f = 0; f < 4; ++f) {
      const _Float16* brow = w + (size_t)(n0 + f * 16 + lo) * DIM_ + hi * 8;
      acc[f] = wmma_f16(af, load_frag_k(brow + k0), acc[f]);
    }
  }
#pragma unroll
  for (int f = 0; f < 4; ++f) {
    const unsigned col = n0 + f * 16 + lo;
    const float bv = bias[col];
#pragma unroll
    for (int r = 0; r < 8; ++r) {
      const unsigned row = m0 + r + 8 * hi;
      out[(size_t)row * DIM_ + col] = acc[f][r] + bv;
    }
  }
}

// ---------------------------------------------------------------------------
// Launch
// ---------------------------------------------------------------------------
extern "C" void kernel_launch(void* const* d_in, const int* in_sizes, int n_in,
                              void* d_out, int out_size, void* d_ws, size_t ws_size,
                              hipStream_t stream) {
  const float* x     = (const float*)d_in[0];
  const float* ln_w  = (const float*)d_in[1];
  const float* ln_b  = (const float*)d_in[2];
  const float* qkv_w = (const float*)d_in[3];
  const float* qkv_b = (const float*)d_in[4];
  const float* out_w = (const float*)d_in[5];
  const float* out_b = (const float*)d_in[6];
  float* out = (float*)d_out;

  char* ws = (char*)d_ws;
  // f16 workspace layout (bytes)
  _Float16* xn   = (_Float16*)(ws + 0);                 // 8192*512*2  = 8 MiB
  _Float16* wq_h = (_Float16*)(ws + 8388608);           // 1536*512*2  = 1.5 MiB
  _Float16* wo_h = (_Float16*)(ws + 9961472);           // 512*512*2   = 0.5 MiB
  _Float16* q_h  = (_Float16*)(ws + 10485760);          // 8 MiB
  _Float16* k_h  = (_Float16*)(ws + 18874368);          // 8 MiB
  _Float16* vt_h = (_Float16*)(ws + 27262976);          // 8 MiB
  _Float16* o_h  = (_Float16*)(ws + 35651584);          // 8 MiB (end 42 MiB)

  cast_weights_k<<<1024, 256, 0, stream>>>(qkv_w, out_w, wq_h, wo_h);
  layernorm_k<<<ROWS, 128, 0, stream>>>(x, ln_w, ln_b, xn);
  qkv_gemm_k<<<dim3(QKVN / 256, ROWS / 32), 256, 0, stream>>>(
      xn, wq_h, qkv_b, q_h, k_h, vt_h);
  attention_k<<<B_ * H_ * (N_ / 128), 256, 0, stream>>>(q_h, k_h, vt_h, o_h);
  out_gemm_k<<<dim3(DIM_ / 256, ROWS / 32), 256, 0, stream>>>(
      o_h, wo_h, out_b, out);
}
